// MemoryModule_54099408060744
// MI455X (gfx1250) — matrix operand order repaired and checked by reading, not compiled, not run on hardware
//
#include <hip/hip_runtime.h>

typedef _Float16 h16 __attribute__((ext_vector_type(16)));
typedef _Float16 h8  __attribute__((ext_vector_type(8)));
typedef float    f8  __attribute__((ext_vector_type(8)));

#define B_   16
#define C_   768
#define N_   4096   // H*W
#define M_   1024
#define EPSN 1e-12f

// ---------------------------------------------------------------------------
// Kernel 1: memory prep.
//   memnH[m][c] = f16( memory[m][c] / max(||memory[m]||, eps) )   (B of GEMM-1)
//   memTH[c][m] = f16( memory[m][c] )                             (B of GEMM-2)
// ---------------------------------------------------------------------------
__global__ __launch_bounds__(256) void k_mem_prep(const float* __restrict__ mem,
                                                  _Float16* __restrict__ memnH,
                                                  _Float16* __restrict__ memTH) {
    __shared__ float srow[C_];
    __shared__ float red[256];
    const int m = blockIdx.x;
    const int tid = threadIdx.x;
    float s = 0.f;
    for (int c = tid; c < C_; c += 256) {
        float v = mem[(size_t)m * C_ + c];
        srow[c] = v;
        s += v * v;
    }
    red[tid] = s;
    __syncthreads();
    for (int off = 128; off > 0; off >>= 1) {
        if (tid < off) red[tid] += red[tid + off];
        __syncthreads();
    }
    const float rn = 1.f / fmaxf(sqrtf(red[0]), EPSN);
    for (int c = tid; c < C_; c += 256) {
        float v = srow[c];
        memnH[(size_t)m * C_ + c] = (_Float16)(v * rn);
        memTH[(size_t)c * M_ + m] = (_Float16)v;
    }
}

// ---------------------------------------------------------------------------
// Kernel 2: per-pixel reciprocal L2 norm of z along C.
// ---------------------------------------------------------------------------
__global__ __launch_bounds__(256) void k_znorm(const float* __restrict__ z,
                                               float* __restrict__ rz) {
    const int idx = blockIdx.x * 256 + threadIdx.x;   // b*N + n
    const int b = idx >> 12;
    const int n = idx & (N_ - 1);
    const float* p = z + (size_t)b * C_ * N_ + n;
    float s = 0.f;
    for (int c = 0; c < C_; ++c) {
        float v = p[(size_t)c * N_];
        s += v * v;
    }
    rz[idx] = 1.f / fmaxf(sqrtf(s), EPSN);
}

// Build a 16x32 A fragment (contiguous k = 16*hi + s) from an LDS row base.
__device__ __forceinline__ h16 load_afrag(const _Float16* ap) {
    h8 alo = *(const h8*)ap;
    h8 ahi = *(const h8*)(ap + 8);
    h16 a;
#pragma unroll
    for (int i = 0; i < 8; ++i) { a[i] = alo[i]; a[i + 8] = ahi[i]; }
    return a;
}

// ---------------------------------------------------------------------------
// Kernel 3: scores = z_norm @ mem_norm^T  -> raw scores into attn region.
// Block = 32 N-rows x all 1024 M (doubles B-operand reuse vs 16 rows).
// 8 waves = 2 row-strips x 4 M-column groups; each wave runs two sequential
// 8-tile M-chunks (acc = 64 VGPRs per chunk). K = 768 (24 WMMA steps).
// ---------------------------------------------------------------------------
#define LDA3 (C_ + 8)   // 1552B row stride -> conflict-free frag reads
__global__ __launch_bounds__(256) void k_scores(const float* __restrict__ z,
                                                const float* __restrict__ rz,
                                                const _Float16* __restrict__ memnH,
                                                float* attn) {
    __shared__ __attribute__((aligned(32))) _Float16 aT[32 * LDA3];   // ~48.5 KB
    const int b  = blockIdx.x >> 7;          // 128 N-tiles (of 32) per batch
    const int n0 = (blockIdx.x & 127) * 32;
    const int tid = threadIdx.x;

    // Stage A strip: 32 rows x 768 ch of normalized z in f16 (r-inner: 128B runs).
    for (int idx = tid; idx < 32 * C_; idx += 256) {
        int c = idx >> 5, r = idx & 31;
        float v = z[((size_t)b * C_ + c) * N_ + n0 + r] * rz[b * N_ + n0 + r];
        aT[r * LDA3 + c] = (_Float16)v;
    }
    __syncthreads();

    const int lane = tid & 31, wave = tid >> 5;
    const int l = lane & 15, hi = lane >> 4;
    const int rbase = (wave & 1) * 16;       // which 16-row strip
    const int wcol  = wave >> 1;             // M column group (0..3)

    const _Float16* aBase = &aT[(rbase + l) * LDA3 + 16 * hi];

    for (int chunk = 0; chunk < 2; ++chunk) {
        const int mt0 = chunk * 32 + wcol * 8;          // first mtile index
        const _Float16* bBase = &memnH[(size_t)(mt0 * 16 + l) * C_ + 16 * hi];

        f8 acc[8] = {};
        for (int ks = 0; ks < 24; ++ks) {
            h16 aF = load_afrag(aBase + ks * 32);
#pragma unroll
            for (int t = 0; t < 8; ++t) {
                h16 bF = *(const h16*)(bBase + (size_t)t * (16 * C_) + ks * 32);
                acc[t] = __builtin_amdgcn_wmma_f32_16x16x32_f16(
                    false, aF, false, bF, (short)0, acc[t], false, false);
            }
        }

        // D layout: lane col = l (m), VGPR r -> row n = rbase + r + 8*hi.
        for (int t = 0; t < 8; ++t) {
            const int m0 = (mt0 + t) * 16;
#pragma unroll
            for (int r = 0; r < 8; ++r) {
                const int n = n0 + rbase + r + 8 * hi;
                attn[((size_t)b * N_ + n) * M_ + m0 + l] = acc[t][r];
            }
        }
    }
}

// ---------------------------------------------------------------------------
// Kernel 4 (fused): stage raw scores -> LDS, softmax rows in LDS (write final
// attn_w to d_out), then z_hat = attn @ memory with WMMA; store in (B,C,N).
// ---------------------------------------------------------------------------
#define LDA5 (M_ + 8)
__global__ __launch_bounds__(256) void k_fused(float* attn,   // raw in / final out
                                               const _Float16* __restrict__ memTH,
                                               float* __restrict__ zhat) {
    __shared__ __attribute__((aligned(32))) _Float16 aT[16 * LDA5];
    __shared__ float red[256];
    const int b  = blockIdx.x >> 8;
    const int n0 = (blockIdx.x & 255) * 16;
    const int tid = threadIdx.x;

    for (int idx = tid; idx < 16 * M_; idx += 256) {
        int r = idx >> 10, m = idx & (M_ - 1);
        aT[r * LDA5 + m] = (_Float16)attn[((size_t)b * N_ + n0 + r) * M_ + m];
    }
    __syncthreads();

    // Softmax per row: 16 threads per row.
    const int rr = tid >> 4, sub = tid & 15;
    float mx = -1e30f;
    for (int m = sub; m < M_; m += 16) mx = fmaxf(mx, (float)aT[rr * LDA5 + m]);
    red[tid] = mx;
    __syncthreads();
    for (int off = 8; off > 0; off >>= 1) {
        if (sub < off) red[tid] = fmaxf(red[tid], red[tid + off]);
        __syncthreads();
    }
    mx = red[rr * 16];
    __syncthreads();
    float s = 0.f;
    for (int m = sub; m < M_; m += 16) {
        float e = __expf((float)aT[rr * LDA5 + m] - mx);
        s += e;
        aT[rr * LDA5 + m] = (_Float16)e;
    }
    red[tid] = s;
    __syncthreads();
    for (int off = 8; off > 0; off >>= 1) {
        if (sub < off) red[tid] += red[tid + off];
        __syncthreads();
    }
    const float inv = 1.f / red[rr * 16];
    __syncthreads();
    for (int m = sub; m < M_; m += 16) {
        float w = (float)aT[rr * LDA5 + m] * inv;
        aT[rr * LDA5 + m] = (_Float16)w;
        attn[((size_t)b * N_ + n0 + rr) * M_ + m] = w;   // final attn_w output
    }
    __syncthreads();

    // GEMM-2: z_hat[n][c] = sum_m W[n][m] * memory[m][c]. K=1024 (32 steps).
    const int lane = tid & 31, wave = tid >> 5;
    const int l = lane & 15, hi = lane >> 4;

    const _Float16* aBase = &aT[l * LDA5 + 16 * hi];
    const _Float16* bBase = &memTH[(size_t)(wave * 96 + l) * M_ + 16 * hi];

    f8 acc[6] = {};
    for (int ks = 0; ks < 32; ++ks) {
        h16 aF = load_afrag(aBase + ks * 32);
#pragma unroll
        for (int t = 0; t < 6; ++t) {
            h16 bF = *(const h16*)(bBase + (size_t)t * (16 * M_) + ks * 32);
            acc[t] = __builtin_amdgcn_wmma_f32_16x16x32_f16(
                false, aF, false, bF, (short)0, acc[t], false, false);
        }
    }

    // Store: lane col c fixed, VGPR r -> n = n0 + r + 8*hi: 8 contiguous floats.
    for (int t = 0; t < 6; ++t) {
        const int c = (wave * 6 + t) * 16 + l;
        float* o = zhat + ((size_t)b * C_ + c) * N_ + n0 + 8 * hi;
        *(f8*)o = acc[t];
    }
}

// ---------------------------------------------------------------------------
extern "C" void kernel_launch(void* const* d_in, const int* in_sizes, int n_in,
                              void* d_out, int out_size, void* d_ws, size_t ws_size,
                              hipStream_t stream) {
    const float* z   = (const float*)d_in[0];   // (B, C, H, W)
    const float* mem = (const float*)d_in[1];   // (M, C)

    float* zhat = (float*)d_out;                          // B*C*N floats
    float* attn = (float*)d_out + (size_t)B_ * C_ * N_;   // B*N*M floats

    char* ws = (char*)d_ws;
    _Float16* memnH = (_Float16*)ws;                            // M*C f16
    _Float16* memTH = (_Float16*)(ws + (size_t)M_ * C_ * 2);    // C*M f16
    float*    rz    = (float*)   (ws + (size_t)M_ * C_ * 4);    // B*N f32

    k_mem_prep<<<M_, 256, 0, stream>>>(mem, memnH, memTH);
    k_znorm<<<(B_ * N_) / 256, 256, 0, stream>>>(z, rz);
    k_scores<<<B_ * (N_ / 32), 256, 0, stream>>>(z, rz, memnH, attn);
    k_fused<<<B_ * (N_ / 16), 256, 0, stream>>>(attn, memTH, zhat);
}
